// SpectralGPT_1382979469663
// MI455X (gfx1250) — matrix operand order repaired and checked by reading, not compiled, not run on hardware
//
#include <hip/hip_runtime.h>
#include <hip/hip_bf16.h>
#include <math.h>

// ---------------------------------------------------------------------------
// SpectralGPT forward for gfx1250 (MI455X). wave32, WMMA f32_16x16x32_f16.
// ---------------------------------------------------------------------------

typedef _Float16 f16;
typedef __attribute__((ext_vector_type(8)))  f16   v8h;
typedef __attribute__((ext_vector_type(16))) f16   v16h;
typedef __attribute__((ext_vector_type(8)))  float v8f;

#define Bn    4
#define Tn    1024
#define Cn    768
#define NHn   12
#define HSn   64
#define NLn   6
#define VOCn  128
#define NEGINF (-3.4028234663852886e38f)

static __device__ __forceinline__ int imin(int a, int b) { return a < b ? a : b; }

static __device__ __forceinline__ v16h frag_cat(const f16* p0, const f16* p1) {
  union { v16h v; v8h h[2]; } u;
  u.h[0] = *(const v8h*)p0;
  u.h[1] = *(const v8h*)p1;
  return u.v;
}
static __device__ __forceinline__ v8f wmma_f16f32(v16h a, v16h b, v8f c) {
  // D = A(16x32 f16) x B(32x16 f16) + C(16x16 f32)
  return __builtin_amdgcn_wmma_f32_16x16x32_f16(false, a, false, b, (short)0, c, false, false);
}
static __device__ __forceinline__ v8f vzero8() {
  v8f z; for (int i = 0; i < 8; i++) z[i] = 0.0f; return z;
}
static __device__ __forceinline__ unsigned pack_h2(float x, float y) {
  union { f16 h[2]; unsigned u; } p;
  p.h[0] = (f16)x; p.h[1] = (f16)y;
  return p.u;
}

// ---------------------------------------------------------------------------
// Embedding: x[b,t,c] = tok_emb[idx[b,t],c] + sum_p vlut[t,p]*pe_w[p,c] + pe_b[c]
// ---------------------------------------------------------------------------
__global__ __launch_bounds__(256) void embed_kernel(
    const int* __restrict__ idx, const float* __restrict__ tok,
    const float* __restrict__ pw, const float* __restrict__ pb,
    const float* __restrict__ vlut, float* __restrict__ x) {
  const int bt = blockIdx.x;
  const int t  = bt & (Tn - 1);
  const int token = idx[bt];
  for (int c = threadIdx.x; c < Cn; c += 256) {
    float acc = pb[c];
    #pragma unroll
    for (int p = 0; p < 10; p++) acc += vlut[t * 10 + p] * pw[p * Cn + c];
    x[(size_t)bt * Cn + c] = tok[(size_t)token * Cn + c] + acc;
  }
}

// ---------------------------------------------------------------------------
// LayerNorm over C=768, one block per row.
// ---------------------------------------------------------------------------
__global__ __launch_bounds__(256) void ln_kernel(
    const float* __restrict__ x, const float* __restrict__ g,
    const float* __restrict__ be, float* __restrict__ out) {
  __shared__ float r1[256];
  __shared__ float r2[256];
  const int row = blockIdx.x;
  const float* xr = x + (size_t)row * Cn;
  float s = 0.f, s2 = 0.f;
  for (int c = threadIdx.x; c < Cn; c += 256) { float v = xr[c]; s += v; s2 += v * v; }
  r1[threadIdx.x] = s; r2[threadIdx.x] = s2;
  __syncthreads();
  for (int st = 128; st > 0; st >>= 1) {
    if (threadIdx.x < st) { r1[threadIdx.x] += r1[threadIdx.x + st]; r2[threadIdx.x] += r2[threadIdx.x + st]; }
    __syncthreads();
  }
  const float mean = r1[0] * (1.0f / Cn);
  const float var  = r2[0] * (1.0f / Cn) - mean * mean;
  const float rstd = rsqrtf(var + 1e-5f);
  for (int c = threadIdx.x; c < Cn; c += 256)
    out[(size_t)row * Cn + c] = (xr[c] - mean) * rstd * g[c] + be[c];
}

// ---------------------------------------------------------------------------
// Generic WMMA GEMM, templated epilogue + weight addressing.
//   SEG64: B element (k,n) at (n>>6)*(Cn*HSn) + k*HSn + (n&63)  (head-major QKV)
//   else : B element (k,n) at k*ldb + n
//   out element (m,n) at (m>>10)*oB + (m&1023)*oM + (n>>6)*oH + (n&63)*oN
// FLAGS: 1=+bias[n], 2=exact GELU, 4=+res[m*N+n], 8=store f16 (else f32)
// Block = 256 thr (8 waves). Block tile 64(M) x 128(N); wave tile 16x64; K step 32.
// Software-pipelined: global loads for step kb+32 issue before WMMAs of step kb.
// ---------------------------------------------------------------------------
#define SPAD 40
template <int FLAGS, bool SEG64>
__global__ __launch_bounds__(256) void gemm_kernel(
    const float* __restrict__ A, const float* __restrict__ Bw,
    const float* __restrict__ bias, const float* __restrict__ res,
    void* __restrict__ out,
    int N, int K, int lda, int ldb,
    int oB, int oM, int oH, int oN) {
  __shared__ __align__(16) f16 sA[64 * SPAD];
  __shared__ __align__(16) f16 sB[128 * SPAD];
  const int tid  = threadIdx.x;
  const int w    = tid >> 5, lane = tid & 31, lo = lane & 15, hi = lane >> 4;
  const int m0   = blockIdx.x * 64, n0 = blockIdx.y * 128;
  const int wm   = (w & 3) * 16;       // wave row tile within block
  const int wn   = (w >> 2) * 64;      // wave col group within block
  v8f acc[4];
  #pragma unroll
  for (int j = 0; j < 4; j++) acc[j] = vzero8();

  // per-thread staging coordinates (pairs of consecutive k)
  // A tile 64x32 = 1024 pairs -> 4/thread ; B tile 32x128 = 2048 pairs -> 8/thread
  float2 aReg[4];
  float  bReg0[8], bReg1[8];

  auto loadA = [&](int kb) {
    #pragma unroll
    for (int i = 0; i < 4; i++) {
      int fp = i * 256 + tid; int kp = fp & 15, m = fp >> 4;
      aReg[i] = *(const float2*)(A + (size_t)(m0 + m) * lda + kb + kp * 2);
    }
  };
  auto loadB = [&](int kb) {
    #pragma unroll
    for (int i = 0; i < 8; i++) {
      int fp = i * 256 + tid; int n = fp & 127, kp = fp >> 7;
      int gn = n0 + n, kk0 = kb + kp * 2;
      size_t off0, off1;
      if (SEG64) {
        size_t base = (size_t)(gn >> 6) * (Cn * HSn) + (gn & 63);
        off0 = base + (size_t)kk0 * HSn;
        off1 = base + (size_t)(kk0 + 1) * HSn;
      } else {
        off0 = (size_t)kk0 * ldb + gn;
        off1 = off0 + ldb;
      }
      bReg0[i] = Bw[off0];
      bReg1[i] = Bw[off1];
    }
  };
  auto storeLDS = [&]() {
    #pragma unroll
    for (int i = 0; i < 4; i++) {
      int fp = i * 256 + tid; int kp = fp & 15, m = fp >> 4;
      *(unsigned*)&sA[m * SPAD + kp * 2] = pack_h2(aReg[i].x, aReg[i].y);
    }
    #pragma unroll
    for (int i = 0; i < 8; i++) {
      int fp = i * 256 + tid; int n = fp & 127, kp = fp >> 7;
      *(unsigned*)&sB[n * SPAD + kp * 2] = pack_h2(bReg0[i], bReg1[i]);
    }
  };

  loadA(0); loadB(0);
  for (int kb = 0; kb < K; kb += 32) {
    storeLDS();
    __syncthreads();
    if (kb + 32 < K) { loadA(kb + 32); loadB(kb + 32); }   // hide VMEM under WMMA
    // A-frag: lane lo = row, k = hi*16+e
    v16h af = frag_cat(&sA[(wm + lo) * SPAD + hi * 16],
                       &sA[(wm + lo) * SPAD + hi * 16 + 8]);
    // preload all B-frags, then issue WMMAs back-to-back
    v16h bf[4];
    #pragma unroll
    for (int j = 0; j < 4; j++) {
      int nn = wn + j * 16 + lo;
      bf[j] = frag_cat(&sB[nn * SPAD + hi * 16], &sB[nn * SPAD + hi * 16 + 8]);
    }
    #pragma unroll
    for (int j = 0; j < 4; j++) acc[j] = wmma_f16f32(af, bf[j], acc[j]);
    __syncthreads();
  }

  // epilogue: C/D layout -> element (row v+8*hi, col lo)
  #pragma unroll
  for (int j = 0; j < 4; j++) {
    #pragma unroll
    for (int v = 0; v < 8; v++) {
      int m = m0 + wm + v + 8 * hi;
      int n = n0 + wn + j * 16 + lo;
      float val = acc[j][v];
      if (FLAGS & 1) val += bias[n];
      if (FLAGS & 2) val = 0.5f * val * (1.0f + erff(val * 0.70710678118654752f));
      if (FLAGS & 4) val += res[(size_t)m * N + n];
      size_t off = (size_t)(m >> 10) * oB + (size_t)(m & 1023) * oM +
                   (size_t)(n >> 6) * oH + (size_t)(n & 63) * oN;
      if (FLAGS & 8) ((f16*)out)[off] = (f16)val;
      else           ((float*)out)[off] = val;
    }
  }
}

// ---------------------------------------------------------------------------
// Attention: one block per (b, head, 16-row Q tile). 8 waves.
//  Phase 1: S = Q K^T * scale + rbias, masked (allowed derived from rbias/causal),
//           S (16 x t0+16, f32) in LDS. Waves split the s-range in 128-chunks.
//  Phase 2: exact softmax per row (2 rows/wave), P (f16) to LDS.
//  Phase 3: O = P x V via WMMA (V stored transposed [d][t]); cross-wave reduce.
// dynamic LDS: 64KB S (reused for partials) + 32KB P = 96KB (<320KB/WGP).
// ---------------------------------------------------------------------------
__global__ __launch_bounds__(256) void attn_kernel(
    const f16* __restrict__ q, const f16* __restrict__ k,
    const f16* __restrict__ vt, const float* __restrict__ rbias,
    float* __restrict__ att_o) {
  extern __shared__ char smem[];
  float* S    = (float*)smem;                      // [16][1024] f32
  f16*   P    = (f16*)(smem + 16 * 1024 * 4);      // [16][1024] f16
  float* part = (float*)smem;                      // reuse S region: [8][16][64]

  const int tid = threadIdx.x, w = tid >> 5, lane = tid & 31, lo = lane & 15, hi = lane >> 4;
  const int t0 = blockIdx.x * 16, hh = blockIdx.y, b = blockIdx.z;
  const int slimit = t0 + 16;                      // causal: s <= t0+15
  const float scale = 0.036084391824351615f;       // C^-0.5 (reference uses C, not HS)

  const f16* qb = q  + (size_t)(b * NHn + hh) * Tn * HSn;
  const f16* kb = k  + (size_t)(b * NHn + hh) * Tn * HSn;
  const f16* vb = vt + (size_t)(b * NHn + hh) * HSn * Tn;

  // Q A-frags for both 32-wide k-steps of HS=64
  v16h q0f = frag_cat(qb + (t0 + lo) * HSn + hi * 16,      qb + (t0 + lo) * HSn + hi * 16 + 8);
  v16h q1f = frag_cat(qb + (t0 + lo) * HSn + 32 + hi * 16, qb + (t0 + lo) * HSn + 32 + hi * 16 + 8);

  // ---- Phase 1: scores ----
  const int send = imin(slimit, w * 128 + 128);
  for (int st = w * 128; st < send; st += 16) {
    // B-frags from row-major K rows (B operand of Q*K^T): lane lo = s-col, k = hi*16+e
    v16h k0f = frag_cat(kb + (st + lo) * HSn + hi * 16,      kb + (st + lo) * HSn + hi * 16 + 8);
    v16h k1f = frag_cat(kb + (st + lo) * HSn + 32 + hi * 16, kb + (st + lo) * HSn + 32 + hi * 16 + 8);
    v8f acc = vzero8();
    acc = wmma_f16f32(q0f, k0f, acc);
    acc = wmma_f16f32(q1f, k1f, acc);
    const int s = st + lo;
    #pragma unroll
    for (int v = 0; v < 8; v++) {
      int r = v + 8 * hi, t = t0 + r;
      float rb = rbias[(size_t)t * Tn + s];
      bool ok = (s <= t) && (rb > 0.0f || s == t);   // reconstructs 'allowed'
      S[r * Tn + s] = ok ? acc[v] * scale + rb : NEGINF;
    }
  }
  __syncthreads();

  // ---- Phase 2: softmax (rows 2w, 2w+1) ----
  const int spad = (slimit + 31) & ~31;
  for (int rr = 0; rr < 2; rr++) {
    int r = w * 2 + rr;
    float mx = NEGINF;
    for (int s = lane; s < slimit; s += 32) mx = fmaxf(mx, S[r * Tn + s]);
    for (int off = 16; off > 0; off >>= 1) mx = fmaxf(mx, __shfl_xor(mx, off, 32));
    float sum = 0.f;
    for (int s = lane; s < slimit; s += 32) sum += __expf(S[r * Tn + s] - mx);
    for (int off = 16; off > 0; off >>= 1) sum += __shfl_xor(sum, off, 32);
    float inv = 1.0f / sum;
    for (int s = lane; s < spad; s += 32) {
      float pv = (s < slimit) ? __expf(S[r * Tn + s] - mx) * inv : 0.0f;
      P[r * Tn + s] = (f16)pv;
    }
  }
  __syncthreads();   // P complete; S region now dead -> reuse for partials

  // ---- Phase 3: O = P x V ----
  v8f o[4];
  #pragma unroll
  for (int j = 0; j < 4; j++) o[j] = vzero8();
  for (int sc = w * 128; sc < send; sc += 32) {
    // A-frag from P rows (16 x 32)
    v16h pf = frag_cat(&P[lo * Tn + sc + hi * 16], &P[lo * Tn + sc + hi * 16 + 8]);
    // preload all V B-frags, then WMMAs back-to-back
    v16h vf[4];
    #pragma unroll
    for (int j = 0; j < 4; j++)
      vf[j] = frag_cat(vb + (j * 16 + lo) * Tn + sc + hi * 16,
                       vb + (j * 16 + lo) * Tn + sc + hi * 16 + 8);
    #pragma unroll
    for (int j = 0; j < 4; j++) o[j] = wmma_f16f32(pf, vf[j], o[j]);
  }
  #pragma unroll
  for (int j = 0; j < 4; j++)
    #pragma unroll
    for (int v = 0; v < 8; v++)
      part[w * 1024 + (v + 8 * hi) * 64 + j * 16 + lo] = o[j][v];
  __syncthreads();

  for (int e = tid; e < 1024; e += 256) {
    float sum = 0.f;
    #pragma unroll
    for (int ww = 0; ww < 8; ww++) sum += part[ww * 1024 + e];
    int r = e >> 6, d = e & 63;
    att_o[((size_t)b * Tn + t0 + r) * Cn + hh * HSn + d] = sum;
  }
}

// ---------------------------------------------------------------------------
// Orchestration
// ---------------------------------------------------------------------------
extern "C" void kernel_launch(void* const* d_in, const int* in_sizes, int n_in,
                              void* d_out, int out_size, void* d_ws, size_t ws_size,
                              hipStream_t stream) {
  (void)in_sizes; (void)n_in; (void)out_size; (void)ws_size;
  const int*   idx     = (const int*)  d_in[0];
  const float* tok_emb = (const float*)d_in[1];
  const float* pe_w    = (const float*)d_in[2];
  const float* pe_b    = (const float*)d_in[3];
  const float* vlut    = (const float*)d_in[4];
  const float* rbias   = (const float*)d_in[5];
  /* d_in[6] 'allowed' (bool) intentionally unused: derived from rbias */
  const float* ln1_g   = (const float*)d_in[7];
  const float* ln1_b   = (const float*)d_in[8];
  const float* qw      = (const float*)d_in[9];
  const float* kw      = (const float*)d_in[10];
  const float* vw      = (const float*)d_in[11];
  const float* proj_w  = (const float*)d_in[12];
  const float* proj_b  = (const float*)d_in[13];
  const float* ln2_g   = (const float*)d_in[14];
  const float* ln2_b   = (const float*)d_in[15];
  const float* ff1_w   = (const float*)d_in[16];
  const float* ff1_b   = (const float*)d_in[17];
  const float* ff2_w   = (const float*)d_in[18];
  const float* ff2_b   = (const float*)d_in[19];
  const float* lnf_g   = (const float*)d_in[20];
  const float* lnf_b   = (const float*)d_in[21];
  const float* head_w  = (const float*)d_in[22];
  const float* head_b  = (const float*)d_in[23];

  char* ws = (char*)d_ws;
  float* x     = (float*)(ws + 0);          // 12 MB  [B,T,C] f32
  float* hbuf  = (float*)(ws + 12582912);   // 12 MB  ln output
  f16*   q     = (f16*)  (ws + 25165824);   // 6 MB   [B,NH,T,HS] f16
  f16*   kk    = (f16*)  (ws + 31457280);   // 6 MB   [B,NH,T,HS] f16
  f16*   vt    = (f16*)  (ws + 37748736);   // 6 MB   [B,NH,HS,T] f16 (transposed)
  float* att_o = (float*)(ws + 44040192);   // 12 MB  [B,T,C] f32
  float* ffmid = (float*)(ws + 56623104);   // 48 MB  [B,T,4C] f32

  const int M = Bn * Tn;                    // 4096
  embed_kernel<<<M, 256, 0, stream>>>(idx, tok_emb, pe_w, pe_b, vlut, x);

  const dim3 gC (M / 64, Cn / 128);         // N=768
  const dim3 gF (M / 64, (4 * Cn) / 128);   // N=3072
  const dim3 gH (M / 64, 1);                // N=128
  const dim3 gA (Tn / 16, NHn, Bn);

  for (int l = 0; l < NLn; l++) {
    const size_t wqkv = (size_t)l * NHn * Cn * HSn;   // 589824
    const size_t wcc  = (size_t)l * Cn * Cn;          // 589824
    const size_t wff  = (size_t)l * Cn * 4 * Cn;      // 2359296

    ln_kernel<<<M, 256, 0, stream>>>(x, ln1_g + l * Cn, ln1_b + l * Cn, hbuf);

    // QKV: head-major weights (SEG64); out f16
    gemm_kernel<8, true><<<gC, 256, 0, stream>>>(hbuf, qw + wqkv, nullptr, nullptr, q,
        Cn, Cn, Cn, HSn,  786432, 64, 65536, 1);
    gemm_kernel<8, true><<<gC, 256, 0, stream>>>(hbuf, kw + wqkv, nullptr, nullptr, kk,
        Cn, Cn, Cn, HSn,  786432, 64, 65536, 1);
    // V written transposed: [B,NH,HS,T]
    gemm_kernel<8, true><<<gC, 256, 0, stream>>>(hbuf, vw + wqkv, nullptr, nullptr, vt,
        Cn, Cn, Cn, HSn,  786432, 1, 65536, 1024);

    attn_kernel<<<gA, 256, 98304, stream>>>(q, kk, vt, rbias, att_o);

    // proj: +bias +residual(x) -> x
    gemm_kernel<1 | 4, false><<<gC, 256, 0, stream>>>(att_o, proj_w + wcc,
        proj_b + l * Cn, x, x,
        Cn, Cn, Cn, Cn,  786432, Cn, 64, 1);

    ln_kernel<<<M, 256, 0, stream>>>(x, ln2_g + l * Cn, ln2_b + l * Cn, hbuf);

    // ff1: +bias +GELU -> ffmid
    gemm_kernel<1 | 2, false><<<gF, 256, 0, stream>>>(hbuf, ff1_w + wff,
        ff1_b + (size_t)l * 4 * Cn, nullptr, ffmid,
        4 * Cn, Cn, Cn, 4 * Cn,  1024 * 4 * Cn, 4 * Cn, 64, 1);

    // ff2: +bias +residual(x) -> x
    gemm_kernel<1 | 4, false><<<gC, 256, 0, stream>>>(ffmid, ff2_w + wff,
        ff2_b + l * Cn, x, x,
        Cn, 4 * Cn, 4 * Cn, Cn,  786432, Cn, 64, 1);
  }

  ln_kernel<<<M, 256, 0, stream>>>(x, lnf_g, lnf_b, hbuf);

  // head: +bias -> d_out (f32, [B,T,VOCAB])
  gemm_kernel<1, false><<<gH, 256, 0, stream>>>(hbuf, head_w, head_b, nullptr, d_out,
      VOCn, Cn, Cn, VOCn,  1024 * VOCn, VOCn, 64, 1);
}
